// CAMModule_64029372448972
// MI455X (gfx1250) — matrix operand order repaired and checked by reading, not compiled, not run on hardware
//
#include <hip/hip_runtime.h>
#include <hip/hip_bf16.h>

#define B_ 16
#define C_ 512
#define N_ 4096   // 64*64

typedef __bf16 bf16;
typedef __attribute__((ext_vector_type(4)))  bf16  v4bf;
typedef __attribute__((ext_vector_type(8)))  bf16  v8bf;
typedef __attribute__((ext_vector_type(16))) bf16  v16bf;
typedef __attribute__((ext_vector_type(8)))  float v8f;

// ---------------------------------------------------------------- async copy (CDNA5)

#if defined(__has_builtin)
#if __has_builtin(__builtin_amdgcn_global_load_async_to_lds_b128)
#define ASYNC_LDS 1
#endif
#endif

// pointer types matching the builtin's signature:
//   param0: v4i in AS(1)  (printed by clang as "int __vector(4) __device__ *")
//   param1: v4i in AS(3)  (LDS)
typedef int v4i __attribute__((vector_size(16)));
typedef __attribute__((address_space(1))) v4i v4i_g;
typedef __attribute__((address_space(3))) v4i v4i_l;

// copy 16 bytes global -> LDS
__device__ __forceinline__ void cp16(bf16* l, const bf16* g) {
#ifdef ASYNC_LDS
  __builtin_amdgcn_global_load_async_to_lds_b128(
      (v4i_g*)(unsigned long long)g,
      (v4i_l*)(unsigned long long)l,
      /*offset=*/0, /*cpol=*/0);
#else
  *reinterpret_cast<uint4*>(l) = *reinterpret_cast<const uint4*>(g);
#endif
}

__device__ __forceinline__ void wait_async_cp() {
#ifdef ASYNC_LDS
#if __has_builtin(__builtin_amdgcn_s_wait_asynccnt)
  __builtin_amdgcn_s_wait_asynccnt(0);
#else
  asm volatile("s_wait_asynccnt 0x0" ::: "memory");
#endif
#endif
}

// ---------------------------------------------------------------- WMMA helpers

__device__ __forceinline__ v8f wmma_bf16(v16bf a, v16bf b, v8f c) {
  // D = A(16x32 bf16) * B(32x16 bf16) + C(16x16 f32)
  return __builtin_amdgcn_wmma_f32_16x16x32_bf16(
      /*neg_a=*/false, a, /*neg_b=*/false, b,
      /*c_mod=*/(short)0, c, /*reuse_a=*/false, /*reuse_b=*/false);
}

__device__ __forceinline__ v16bf frag_cat(v8bf lo, v8bf hi) {
  return __builtin_shufflevector(lo, hi, 0, 1, 2, 3, 4, 5, 6, 7,
                                 8, 9, 10, 11, 12, 13, 14, 15);
}

// A fragment (16x32, bf16): lane holds row (lane&15).
// elems 0..7  = K[ph*8 .. +7], elems 8..15 = K[16+ph*8 .. +7], ph = lane>>4.
__device__ __forceinline__ v16bf load_a_frag(const bf16* row_ptr, int ph) {
  v8bf lo = *reinterpret_cast<const v8bf*>(row_ptr + ph * 8);
  v8bf hi = *reinterpret_cast<const v8bf*>(row_ptr + 16 + ph * 8);
  return frag_cat(lo, hi);
}

// B fragment (32x16, bf16): lane holds column (lane&15),
// elems 0..7  = K[ph*8 .. +7], elems 8..15 = K[16+ph*8 .. +7], ph = lane>>4.
__device__ __forceinline__ v16bf load_b_frag(const bf16* col_ptr, int ph) {
  v8bf lo = *reinterpret_cast<const v8bf*>(col_ptr + ph * 8);
  v8bf hi = *reinterpret_cast<const v8bf*>(col_ptr + 16 + ph * 8);
  return frag_cat(lo, hi);
}

// ---------------------------------------------------------------- kernel 1: f32 -> bf16

__global__ void __launch_bounds__(256)
k_cvt_bf16(const float* __restrict__ x, __hip_bfloat16* __restrict__ outp, int n4) {
  int i = blockIdx.x * blockDim.x + threadIdx.x;
  if (i >= n4) return;
  const float4 f = reinterpret_cast<const float4*>(x)[i];
  v4bf o;
  o[0] = (bf16)f.x; o[1] = (bf16)f.y; o[2] = (bf16)f.z; o[3] = (bf16)f.w;
  reinterpret_cast<v4bf*>(outp)[i] = o;
}

// ---------------------------------------------------------------- kernel 2: energy = Xf * Xf^T
// 128x128 tile per block; 8 waves (4 row-groups x 2 col-groups), 8 WMMA accums each.
// Double-buffered LDS staging via async global->LDS copies.

__global__ void __launch_bounds__(256)
k_energy(const __hip_bfloat16* __restrict__ xbp, float* __restrict__ energy) {
  __shared__ bf16 lA[2][128][40];   // rows i0.., 32 K, padded to 40 (bank-spread)
  __shared__ bf16 lB[2][128][40];   // rows j0.. (columns of energy)
  const bf16* xb = reinterpret_cast<const bf16*>(xbp);

  const int b  = blockIdx.z;
  const int i0 = blockIdx.y * 128;
  const int j0 = blockIdx.x * 128;
  const int t    = threadIdx.x;
  const int lane = t & 31, wave = t >> 5;
  const int wi = wave >> 1;            // 0..3 : 32 rows each
  const int wj = wave & 1;             // 0..1 : 64 cols each
  const int ph = lane >> 4;

  const bf16* xbB = xb + (size_t)b * C_ * N_;
  const int lrow = t >> 1;             // 0..127
  const int lcol = (t & 1) * 16;       // 0 or 16 elements
  const bf16* gA = xbB + (size_t)(i0 + lrow) * N_ + lcol;
  const bf16* gB = xbB + (size_t)(j0 + lrow) * N_ + lcol;

  auto stage = [&](int p, int k0) {
    cp16(&lA[p][lrow][lcol],     gA + k0);
    cp16(&lA[p][lrow][lcol + 8], gA + k0 + 8);
    cp16(&lB[p][lrow][lcol],     gB + k0);
    cp16(&lB[p][lrow][lcol + 8], gB + k0 + 8);
  };

  v8f c[2][4];
#pragma unroll
  for (int ti = 0; ti < 2; ++ti)
#pragma unroll
    for (int tj = 0; tj < 4; ++tj)
#pragma unroll
      for (int r = 0; r < 8; ++r) c[ti][tj][r] = 0.0f;

  const int nsteps = N_ / 32;
  stage(0, 0);                          // prologue

  for (int i = 0; i < nsteps; ++i) {
    const int p = i & 1;
    wait_async_cp();                    // buffer p landed in LDS
    __syncthreads();                    // ...and everyone is done reading p^1
    if (i + 1 < nsteps) stage(p ^ 1, (i + 1) * 32);

    v16bf af[2], bfrag[4];
#pragma unroll
    for (int ti = 0; ti < 2; ++ti)
      af[ti] = load_a_frag(&lA[p][wi * 32 + ti * 16 + (lane & 15)][0], ph);
#pragma unroll
    for (int tj = 0; tj < 4; ++tj)
      bfrag[tj] = load_b_frag(&lB[p][wj * 64 + tj * 16 + (lane & 15)][0], ph);
#pragma unroll
    for (int ti = 0; ti < 2; ++ti)
#pragma unroll
      for (int tj = 0; tj < 4; ++tj)
        c[ti][tj] = wmma_bf16(af[ti], bfrag[tj], c[ti][tj]);
  }

  // C/D layout: VGPR r holds row (ph*8 + r); lanes 0..15 hold the 16 columns.
#pragma unroll
  for (int ti = 0; ti < 2; ++ti) {
#pragma unroll
    for (int tj = 0; tj < 4; ++tj) {
      const int row0 = i0 + wi * 32 + ti * 16 + ph * 8;
      const int col  = j0 + wj * 64 + tj * 16 + (lane & 15);
      float* e = energy + ((size_t)b * C_ + row0) * C_ + col;
#pragma unroll
      for (int r = 0; r < 8; ++r) e[(size_t)r * C_] = c[ti][tj][r];
    }
  }
}

// ---------------------------------------------------------------- kernel 3: softmax(max - e)
// softmax(max_j e - e_j) == softmax(-e_j); stabilize with row MIN:
// p_j = exp(min_j e - e_j) / sum. One 256-thread block per row (512 elems).

__global__ void __launch_bounds__(256)
k_softmax(const float* __restrict__ energy, __hip_bfloat16* __restrict__ attnp) {
  bf16* attn = reinterpret_cast<bf16*>(attnp);
  const int row = blockIdx.x;            // b*C + i
  const float* e = energy + (size_t)row * C_;
  const int t = threadIdx.x;
  const int lane = t & 31, wave = t >> 5;

  float e0 = e[t], e1 = e[t + 256];
  float m = fminf(e0, e1);
#pragma unroll
  for (int off = 16; off >= 1; off >>= 1) m = fminf(m, __shfl_xor(m, off, 32));

  __shared__ float rmin[8], rsum[8];
  if (lane == 0) rmin[wave] = m;
  __syncthreads();
  m = rmin[0];
#pragma unroll
  for (int w = 1; w < 8; ++w) m = fminf(m, rmin[w]);

  float p0 = __expf(m - e0), p1 = __expf(m - e1);
  float s = p0 + p1;
#pragma unroll
  for (int off = 16; off >= 1; off >>= 1) s += __shfl_xor(s, off, 32);
  if (lane == 0) rsum[wave] = s;
  __syncthreads();
  s = 0.0f;
#pragma unroll
  for (int w = 0; w < 8; ++w) s += rsum[w];
  const float inv = 1.0f / s;

  bf16* o = attn + (size_t)row * C_;
  o[t]       = (bf16)(p0 * inv);
  o[t + 256] = (bf16)(p1 * inv);
}

// ---------------------------------------------------------------- kernel 4: y = gamma * (attn @ Xf) + x
// 128 (rows) x 64 (cols of N) tile per block; A staged async, B staged transposed.

__global__ void __launch_bounds__(256)
k_out(const __hip_bfloat16* __restrict__ attnp, const __hip_bfloat16* __restrict__ xbp,
      const float* __restrict__ x, const float* __restrict__ gamma,
      float* __restrict__ y) {
  __shared__ bf16 lA[2][128][40];    // attn rows, 32 K (over j)
  __shared__ bf16 lBt[2][64][40];    // transposed: [n_local][k_local]
  const bf16* attn = reinterpret_cast<const bf16*>(attnp);
  const bf16* xb   = reinterpret_cast<const bf16*>(xbp);

  const int b  = blockIdx.z;
  const int i0 = blockIdx.y * 128;
  const int n0 = blockIdx.x * 64;
  const int t    = threadIdx.x;
  const int lane = t & 31, wave = t >> 5;
  const int wi = wave >> 1;            // 0..3 : 32 rows each
  const int wj = wave & 1;             // 0..1 : 32 cols each
  const int ph = lane >> 4;

  const bf16* aB  = attn + (size_t)b * C_ * C_;
  const bf16* xbB = xb + (size_t)b * C_ * N_;

  const int lrow = t >> 1;             // 0..127 (A loader)
  const int lcol = (t & 1) * 16;
  const bf16* gA = aB + (size_t)(i0 + lrow) * C_ + lcol;
  const int kk = t >> 3;               // 0..31  (B loader: k row)
  const int nn = (t & 7) * 8;          // 0..56  (8 contiguous n per thread)
  const bf16* gB = xbB + (size_t)kk * N_ + n0 + nn;

  auto stage = [&](int p, int k0) {
    cp16(&lA[p][lrow][lcol],     gA + k0);
    cp16(&lA[p][lrow][lcol + 8], gA + k0 + 8);
    // coalesced read along n, transpose into LDS so B-frag reads are contiguous in K
    v8bf q = *reinterpret_cast<const v8bf*>(gB + (size_t)k0 * N_);
#pragma unroll
    for (int u = 0; u < 8; ++u) lBt[p][nn + u][kk] = q[u];
  };

  v8f c[2][2];
#pragma unroll
  for (int ti = 0; ti < 2; ++ti)
#pragma unroll
    for (int tj = 0; tj < 2; ++tj)
#pragma unroll
      for (int r = 0; r < 8; ++r) c[ti][tj][r] = 0.0f;

  const int nsteps = C_ / 32;
  stage(0, 0);                          // prologue

  for (int i = 0; i < nsteps; ++i) {
    const int p = i & 1;
    wait_async_cp();
    __syncthreads();
    if (i + 1 < nsteps) stage(p ^ 1, (i + 1) * 32);

    v16bf af[2], bfrag[2];
#pragma unroll
    for (int ti = 0; ti < 2; ++ti)
      af[ti] = load_a_frag(&lA[p][wi * 32 + ti * 16 + (lane & 15)][0], ph);
#pragma unroll
    for (int tj = 0; tj < 2; ++tj)
      bfrag[tj] = load_b_frag(&lBt[p][wj * 32 + tj * 16 + (lane & 15)][0], ph);
#pragma unroll
    for (int ti = 0; ti < 2; ++ti)
#pragma unroll
      for (int tj = 0; tj < 2; ++tj)
        c[ti][tj] = wmma_bf16(af[ti], bfrag[tj], c[ti][tj]);
  }

  const float g = gamma[0];
#pragma unroll
  for (int ti = 0; ti < 2; ++ti) {
#pragma unroll
    for (int tj = 0; tj < 2; ++tj) {
      const int row0 = i0 + wi * 32 + ti * 16 + ph * 8;
      const int col  = n0 + wj * 32 + tj * 16 + (lane & 15);
      const size_t base = ((size_t)b * C_ + row0) * N_ + col;
#pragma unroll
      for (int r = 0; r < 8; ++r)
        y[base + (size_t)r * N_] = g * c[ti][tj][r] + x[base + (size_t)r * N_];
    }
  }
}

// ---------------------------------------------------------------- launch

extern "C" void kernel_launch(void* const* d_in, const int* in_sizes, int n_in,
                              void* d_out, int out_size, void* d_ws, size_t ws_size,
                              hipStream_t stream) {
  const float* x     = (const float*)d_in[0];
  const float* gamma = (const float*)d_in[1];
  float* y = (float*)d_out;

  // Workspace layout (88 MiB total):
  //   [0,      64 MiB) : xb   bf16 [B, C, N]
  //   [64 MiB, 80 MiB) : energy f32 [B, C, C]
  //   [80 MiB, 88 MiB) : attn bf16 [B, C, C]
  char* w = (char*)d_ws;
  __hip_bfloat16* xb     = (__hip_bfloat16*)(w);
  float*          energy = (float*)(w + ((size_t)64 << 20));
  __hip_bfloat16* attn   = (__hip_bfloat16*)(w + ((size_t)80 << 20));

  const int total = B_ * C_ * N_;      // 33,554,432
  k_cvt_bf16<<<dim3(total / 1024), dim3(256), 0, stream>>>(x, xb, total / 4);
  k_energy  <<<dim3(C_ / 128, C_ / 128, B_), dim3(256), 0, stream>>>(xb, energy);
  k_softmax <<<dim3(B_ * C_), dim3(256), 0, stream>>>(energy, attn);
  k_out     <<<dim3(N_ / 64, C_ / 128, B_), dim3(256), 0, stream>>>(attn, xb, x, gamma, y);
}
